// GroupedQueryAttention_61306363183650
// MI455X (gfx1250) — compile-verified, matrix operand order checked by reading, not run on hardware
//
#include <hip/hip_runtime.h>
#include <hip/hip_bf16.h>

typedef _Float16 f16_t;
typedef __attribute__((ext_vector_type(16))) _Float16 v16h;
typedef __attribute__((ext_vector_type(8)))  float    v8f;

#define B_   2
#define S_   2048
#define E_   1024
#define NH_  16
#define NKV_ 4
#define HD_  64
#define KVE_ 256   /* NKV*HD */
#define GS_  4

// K-index striping for 16-bit A/B WMMA fragments (ISA 05_wmma §7.12.2):
// lanes 0-15 hold K = {0..7, 16..23}, lanes 16-31 hold K = {8..15, 24..31}
__device__ __forceinline__ int kmap16(int i, int lane) {
    int lo = (lane & 16) ? 8 : 0;
    return (i < 8) ? (i + lo) : (16 + lo + (i - 8));
}

// LDS byte address = low 32 bits of the generic address (ISA: LDS aperture).
__device__ __forceinline__ uint32_t to_lds(const void* p) {
    return (uint32_t)(uintptr_t)p;
}

// Issue one 16-byte async global->LDS copy (tracked by ASYNCcnt).
__device__ __forceinline__ void async_b128(uint32_t lds_byte, const void* gptr) {
    asm volatile("global_load_async_to_lds_b128 %0, %1, off"
                 :: "v"(lds_byte), "v"((uint64_t)(uintptr_t)gptr)
                 : "memory");
}

#define APAD 36   /* 32 f32 per row + 4 pad (keeps 16B alignment, kills bank conflicts) */

// Y[M,N] = (X[M,K] . W[N,K]^T) * scale  — f16 WMMA, f32 accumulate.
// Block = 128 threads (4 waves) computing a 64x64 output tile:
// wave w owns rows [m0+16w, m0+16w+16) x all 64 cols (4 accumulators).
// A panel (64x32) and B panel (64 rows of W x 32 k) staged into LDS with
// GLOBAL_LOAD_ASYNC_TO_LDS_B128; fragments built from LDS with f32->f16 cvt.
template <typename TIN, typename TOUT>
__global__ __launch_bounds__(128) void gemm_xwt_kernel(
    const TIN* __restrict__ X, const float* __restrict__ W,
    TOUT* __restrict__ Y, int M, int N, int K, float scale)
{
    __shared__ float a32[64][APAD];   // X panel, f32
    __shared__ float b32[64][APAD];   // W panel, f32 (row n, col k)
    const int lane = threadIdx.x & 31;
    const int wave = threadIdx.x >> 5;
    const int m0 = blockIdx.x * 64;
    const int n0 = blockIdx.y * 64;
    const int nw  = lane & 15;
    const int hi8 = (lane & 16) ? 8 : 0;

    v8f acc[4] = {{}, {}, {}, {}};

    for (int k0 = 0; k0 < K; k0 += 32) {
        // Async-stage both panels: 64 rows x 8 x 16B chunks each = 512 chunks,
        // 128 threads -> 4 chunks per panel per thread, fully coalesced b128.
        for (int e = threadIdx.x; e < 512; e += 128) {
            int r = e >> 3, c = (e & 7) * 4;          // row, f32 column
            async_b128(to_lds(&a32[r][c]), &X[(size_t)(m0 + r) * K + k0 + c]);
            async_b128(to_lds(&b32[r][c]), &W[(size_t)(n0 + r) * K + k0 + c]);
        }
        asm volatile("s_wait_asynccnt 0x0" ::: "memory");  // this wave's copies done
        __syncthreads();                                   // all waves' copies done

        v16h a;
        #pragma unroll
        for (int i = 0; i < 16; ++i)
            a[i] = (_Float16)a32[wave * 16 + nw][kmap16(i, lane)];
        #pragma unroll
        for (int j = 0; j < 4; ++j) {
            v16h bf;   // B(k,n) = W[n][k], n = n0 + j*16 + nw
            #pragma unroll
            for (int i = 0; i < 16; ++i)
                bf[i] = (_Float16)b32[j * 16 + nw][kmap16(i, lane)];
            acc[j] = __builtin_amdgcn_wmma_f32_16x16x32_f16(false, a, false, bf,
                                                            (short)0, acc[j], false, false);
        }
        __syncthreads();
    }

    #pragma unroll
    for (int j = 0; j < 4; ++j)
        #pragma unroll
        for (int r = 0; r < 8; ++r)
            Y[(size_t)(m0 + wave * 16 + r + hi8) * N + n0 + j * 16 + nw] =
                (TOUT)(acc[j][r] * scale);
}

// NOTE: X may be f32 (projections) or f16 (output projection). Async copy is a
// raw byte mover, so for f16 input we stage raw f16 rows (half the chunks).
template <>
__global__ __launch_bounds__(128) void gemm_xwt_kernel<_Float16, float>(
    const _Float16* __restrict__ X, const float* __restrict__ W,
    float* __restrict__ Y, int M, int N, int K, float scale)
{
    __shared__ _Float16 a16[64][40];  // X panel, f16 (32 + 8 pad halves per row)
    __shared__ float    b32[64][APAD];
    const int lane = threadIdx.x & 31;
    const int wave = threadIdx.x >> 5;
    const int m0 = blockIdx.x * 64;
    const int n0 = blockIdx.y * 64;
    const int nw  = lane & 15;
    const int hi8 = (lane & 16) ? 8 : 0;

    v8f acc[4] = {{}, {}, {}, {}};

    for (int k0 = 0; k0 < K; k0 += 32) {
        // A panel: 64 rows x 32 f16 = 64B/row = 4 x 16B chunks -> 256 chunks.
        for (int e = threadIdx.x; e < 256; e += 128) {
            int r = e >> 2, c = (e & 3) * 8;          // row, f16 column
            async_b128(to_lds(&a16[r][c]), &X[(size_t)(m0 + r) * K + k0 + c]);
        }
        // B panel: 512 x 16B chunks.
        for (int e = threadIdx.x; e < 512; e += 128) {
            int r = e >> 3, c = (e & 7) * 4;
            async_b128(to_lds(&b32[r][c]), &W[(size_t)(n0 + r) * K + k0 + c]);
        }
        asm volatile("s_wait_asynccnt 0x0" ::: "memory");
        __syncthreads();

        v16h a;
        #pragma unroll
        for (int i = 0; i < 16; ++i)
            a[i] = a16[wave * 16 + nw][kmap16(i, lane)];
        #pragma unroll
        for (int j = 0; j < 4; ++j) {
            v16h bf;
            #pragma unroll
            for (int i = 0; i < 16; ++i)
                bf[i] = (_Float16)b32[j * 16 + nw][kmap16(i, lane)];
            acc[j] = __builtin_amdgcn_wmma_f32_16x16x32_f16(false, a, false, bf,
                                                            (short)0, acc[j], false, false);
        }
        __syncthreads();
    }

    #pragma unroll
    for (int j = 0; j < 4; ++j)
        #pragma unroll
        for (int r = 0; r < 8; ++r)
            Y[(size_t)(m0 + wave * 16 + r + hi8) * N + n0 + j * 16 + nw] =
                acc[j][r] * scale;
}

// Flash-style causal GQA core. Block = 128 (4 waves); wave w handles q-tile
// blockIdx.x*4+w of head blockIdx.y, batch blockIdx.z. Q pre-scaled by 1/sqrt(HD).
__global__ __launch_bounds__(128) void gqa_attn_kernel(
    const f16_t* __restrict__ Qb, const f16_t* __restrict__ Kb,
    const f16_t* __restrict__ Vb, f16_t* __restrict__ Ob)
{
    __shared__ _Float16 p_lds[4][16][36];        // per-wave P scratch (16x32 + pad)
    const int lane  = threadIdx.x & 31;
    const int wave  = threadIdx.x >> 5;
    const int qtile = blockIdx.x * 4 + wave;
    const int qh    = blockIdx.y;
    const int bb    = blockIdx.z;
    const int kvh   = qh / GS_;
    const int nw    = lane & 15;
    const int hi8   = (lane & 16) ? 8 : 0;

    // Q tile 16x64 -> two 16x32 A fragments (d 0..31 / 32..63)
    v16h q0, q1;
    {
        const f16_t* qrow = Qb + (size_t)(bb * S_ + qtile * 16 + nw) * E_ + qh * HD_;
        #pragma unroll
        for (int i = 0; i < 16; ++i) {
            int k = kmap16(i, lane);
            q0[i] = qrow[k];
            q1[i] = qrow[32 + k];
        }
    }

    v8f o0 = {}, o1 = {}, o2 = {}, o3 = {};
    float m_run[8], l_run[8];
    #pragma unroll
    for (int r = 0; r < 8; ++r) { m_run[r] = -1e30f; l_run[r] = 0.0f; }

    const int nch = (qtile * 16 + 16 + 31) >> 5;     // causal: chunks of 32 keys
    for (int ch = 0; ch < nch; ++ch) {
        const int kc = ch * 32;

        // scores for two 16-key tiles: S = Q(16x64) . K^T(64x16), 2 WMMA each
        v8f s[2];
        #pragma unroll
        for (int t = 0; t < 2; ++t) {
            v16h bk0, bk1;   // B(k=dim, n=key) = K[key][dim]
            const f16_t* krow =
                Kb + (size_t)(bb * S_ + kc + t * 16 + nw) * KVE_ + kvh * HD_;
            #pragma unroll
            for (int i = 0; i < 16; ++i) {
                int k = kmap16(i, lane);
                bk0[i] = krow[k];
                bk1[i] = krow[32 + k];
            }
            v8f a = {};
            a = __builtin_amdgcn_wmma_f32_16x16x32_f16(false, q0, false, bk0,
                                                       (short)0, a, false, false);
            a = __builtin_amdgcn_wmma_f32_16x16x32_f16(false, q1, false, bk1,
                                                       (short)0, a, false, false);
            s[t] = a;
        }

        // causal mask (C-fragment: row = r + hi8, key col = lane&15)
        float sm0[8], sm1[8];
        #pragma unroll
        for (int r = 0; r < 8; ++r) {
            int qg = qtile * 16 + r + hi8;
            sm0[r] = ((kc + nw)      > qg) ? -1e30f : s[0][r];
            sm1[r] = ((kc + 16 + nw) > qg) ? -1e30f : s[1][r];
        }

        // online softmax: row reductions across the 16 lanes of each half-wave
        #pragma unroll
        for (int r = 0; r < 8; ++r) {
            float v = fmaxf(sm0[r], sm1[r]);
            #pragma unroll
            for (int off = 1; off < 16; off <<= 1)
                v = fmaxf(v, __shfl_xor(v, off, 32));
            float mn    = fmaxf(m_run[r], v);
            float alpha = __expf(m_run[r] - mn);
            m_run[r]    = mn;
            float p0 = __expf(sm0[r] - mn);
            float p1 = __expf(sm1[r] - mn);
            float ps = p0 + p1;
            #pragma unroll
            for (int off = 1; off < 16; off <<= 1)
                ps += __shfl_xor(ps, off, 32);
            l_run[r] = l_run[r] * alpha + ps;
            o0[r] *= alpha; o1[r] *= alpha; o2[r] *= alpha; o3[r] *= alpha;
            // reshuffle P from C-layout to row-major LDS for A-fragment reload
            p_lds[wave][r + hi8][nw]      = (_Float16)p0;
            p_lds[wave][r + hi8][16 + nw] = (_Float16)p1;
        }
        asm volatile("s_wait_dscnt 0x0" ::: "memory");   // in-wave DS stores done

        // P (16x32) as A fragment
        v16h pa;
        #pragma unroll
        for (int i = 0; i < 16; ++i)
            pa[i] = p_lds[wave][nw][kmap16(i, lane)];

        // O += P(16x32) . V(32x64): four B fragments along HD
        #pragma unroll
        for (int j = 0; j < 4; ++j) {
            v16h bv;   // B(k=key_local, n=dim_local) = V[kc+key][kvh*64 + j*16 + dim]
            #pragma unroll
            for (int i = 0; i < 16; ++i) {
                int kl = kmap16(i, lane);
                bv[i] = Vb[(size_t)(bb * S_ + kc + kl) * KVE_ + kvh * HD_ + j * 16 + nw];
            }
            v8f* op = (j == 0) ? &o0 : (j == 1) ? &o1 : (j == 2) ? &o2 : &o3;
            *op = __builtin_amdgcn_wmma_f32_16x16x32_f16(false, pa, false, bv,
                                                         (short)0, *op, false, false);
        }
    }

    // normalize and emit into [B*S, NH*HD] (already the transpose-merged layout)
    #pragma unroll
    for (int r = 0; r < 8; ++r) {
        float inv = 1.0f / l_run[r];
        f16_t* orow = Ob + (size_t)(bb * S_ + qtile * 16 + r + hi8) * E_ + qh * HD_;
        orow[nw]      = (f16_t)(o0[r] * inv);
        orow[16 + nw] = (f16_t)(o1[r] * inv);
        orow[32 + nw] = (f16_t)(o2[r] * inv);
        orow[48 + nw] = (f16_t)(o3[r] * inv);
    }
}

extern "C" void kernel_launch(void* const* d_in, const int* in_sizes, int n_in,
                              void* d_out, int out_size, void* d_ws, size_t ws_size,
                              hipStream_t stream) {
    (void)in_sizes; (void)n_in; (void)out_size; (void)ws_size;
    const float* query = (const float*)d_in[0];
    const float* key_t = (const float*)d_in[1];
    const float* value = (const float*)d_in[2];
    /* d_in[3] = causal mask: implemented analytically, unused */
    const float* Wq = (const float*)d_in[4];
    const float* Wk = (const float*)d_in[5];
    const float* Wv = (const float*)d_in[6];
    const float* Wo = (const float*)d_in[7];
    float* out = (float*)d_out;

    char* ws = (char*)d_ws;
    f16_t* Qb = (f16_t*)(ws);                               //  8 MB: [4096,1024]
    f16_t* Kb = (f16_t*)(ws + (size_t)8  * 1024 * 1024);    //  2 MB: [4096, 256]
    f16_t* Vb = (f16_t*)(ws + (size_t)10 * 1024 * 1024);    //  2 MB: [4096, 256]
    f16_t* Ab = (f16_t*)(ws + (size_t)12 * 1024 * 1024);    //  8 MB: [4096,1024]

    const int M = B_ * S_;
    dim3 blk(128);
    const float qscale = 0.125f;   // 1/sqrt(HD), folded into Q projection

    gemm_xwt_kernel<float, _Float16>
        <<<dim3(M / 64, E_ / 64), blk, 0, stream>>>(query, Wq, Qb, M, E_, E_, qscale);
    gemm_xwt_kernel<float, _Float16>
        <<<dim3(M / 64, KVE_ / 64), blk, 0, stream>>>(key_t, Wk, Kb, M, KVE_, E_, 1.0f);
    gemm_xwt_kernel<float, _Float16>
        <<<dim3(M / 64, KVE_ / 64), blk, 0, stream>>>(value, Wv, Vb, M, KVE_, E_, 1.0f);

    gqa_attn_kernel<<<dim3(S_ / 64, NH_, B_), blk, 0, stream>>>(Qb, Kb, Vb, Ab);

    gemm_xwt_kernel<_Float16, float>
        <<<dim3(M / 64, E_ / 64), blk, 0, stream>>>(Ab, Wo, out, M, E_, E_, 1.0f);
}
